// CascadeReranker_12695923327037
// MI455X (gfx1250) — matrix operand order repaired and checked by reading, not compiled, not run on hardware
//
#include <hip/hip_runtime.h>
#include <hip/hip_bf16.h>
#include <math.h>

// ---------------- model dims ----------------
#define BDIM 64
#define KDIM 128
#define INF_ 16
#define EDIM 128
#define HN   4
#define FFND 512
#define NL_  3
#define NTOK (KDIM*BDIM)   // 8192 tokens, token t = k*B + b  (h is (K,B,E))

typedef __attribute__((ext_vector_type(16))) _Float16 v16h;
typedef __attribute__((ext_vector_type(8)))  float    v8f;
typedef __attribute__((ext_vector_type(4)))  float    f4;
typedef __attribute__((ext_vector_type(4)))  _Float16 h4;

__device__ __forceinline__ float gelu_exact(float x){
  return 0.5f * x * (1.0f + erff(x * 0.7071067811865475f));
}

__device__ __forceinline__ v8f wmma_f16(v16h a, v16h b, v8f c){
  // v_wmma_f32_16x16x32_f16  D = A(16x32) * B(32x16) + C
  return __builtin_amdgcn_wmma_f32_16x16x32_f16(false, a, false, b, (short)0, c, false, false);
}

// CDNA5 async DMA: global -> LDS, 16 bytes per lane, tracked by ASYNCcnt.
__device__ __forceinline__ void async_g2l_b128(void* lds_ptr, const void* gptr){
  unsigned lds_off = (unsigned)(size_t)lds_ptr;              // low 32 bits of generic = LDS byte addr
  unsigned long long ga = (unsigned long long)(size_t)gptr;
  asm volatile("global_load_async_to_lds_b128 %0, %1, off" :: "v"(lds_off), "v"(ga) : "memory");
}
__device__ __forceinline__ void wait_async0(){
  asm volatile("s_wait_asynccnt 0x0" ::: "memory");
}

__device__ __forceinline__ h4 cvt4(f4 v){
  h4 o; o[0]=(_Float16)v[0]; o[1]=(_Float16)v[1]; o[2]=(_Float16)v[2]; o[3]=(_Float16)v[3]; return o;
}

// A fragment: LDS row-major [m][k]. Layout: lanes0-15 M=lane, K={0..15};
// lanes16-31 M=lane-16, K={16..31}.
__device__ __forceinline__ v16h frag_a_rm(const _Float16* base, int ld, int mbase, int kbase){
  int lane = threadIdx.x & 31;
  int hi = lane >> 4;
  const _Float16* row = base + (size_t)(mbase + (lane & 15)) * ld + kbase;
  int kA = hi ? 16 : 0, kB = hi ? 24 : 8;
  v16h a;
#pragma unroll
  for (int t = 0; t < 8; ++t) a[t] = row[kA + t];
#pragma unroll
  for (int t = 0; t < 8; ++t) a[8 + t] = row[kB + t];
  return a;
}

// B fragment when LDS holds W as [n][k]:  B(k,n) = W[n][k].
__device__ __forceinline__ v16h frag_b_nk(const _Float16* base, int ld, int nbase, int kbase){
  int lane = threadIdx.x & 31;
  const _Float16* row = base + (size_t)(nbase + (lane & 15)) * ld + kbase + ((lane >> 4) ? 16 : 0);
  v16h b;
#pragma unroll
  for (int t = 0; t < 16; ++t) b[t] = row[t];
  return b;
}

// B fragment when LDS holds matrix as [k][n] (e.g. V for attn*V).
__device__ __forceinline__ v16h frag_b_kn(const _Float16* base, int ld, int nbase, int kbase){
  int lane = threadIdx.x & 31;
  int k0 = kbase + ((lane >> 4) ? 16 : 0);
  const _Float16* col = base + (nbase + (lane & 15));
  v16h b;
#pragma unroll
  for (int t = 0; t < 16; ++t) b[t] = col[(size_t)(k0 + t) * ld];
  return b;
}

// ---------------- per-particle quantities + sum_pt ----------------
__global__ void k_particle(const float* lv, const unsigned char* mask, float* pp, float* sumpt){
  int b = blockIdx.x, k = threadIdx.x;
  float mv = mask[b*KDIM + k] ? 1.f : 0.f;
  const float* L = lv + (size_t)b*4*KDIM;
  float px = L[0*KDIM+k]*mv, py = L[1*KDIM+k]*mv, pz = L[2*KDIM+k]*mv, en = L[3*KDIM+k]*mv;
  float pt  = sqrtf(px*px + py*py);
  float rap = 0.5f * log1pf(2.f*pz / fmaxf(en - pz, 1e-20f));
  float phi = atan2f(py, px);
  int o = b*KDIM + k;
  pp[0*NTOK+o]=pt; pp[1*NTOK+o]=rap; pp[2*NTOK+o]=phi;
  pp[3*NTOK+o]=px; pp[4*NTOK+o]=py; pp[5*NTOK+o]=pz; pp[6*NTOK+o]=en;
  __shared__ float red[KDIM];
  red[k]=pt; __syncthreads();
  for(int s=64;s>0;s>>=1){ if(k<s) red[k]+=red[k+s]; __syncthreads(); }
  if(k==0) sumpt[b]=fmaxf(red[0],1e-6f);
}

// ---------------- embedding: BN -> LN -> linear(18->128) -> GELU -> *mask ----------------
__global__ void k_embed(const float* feat, const float* s1, const unsigned char* mask,
                        const float* pp, const float* sumpt,
                        const float* bnb, const float* bng, const float* bnm, const float* bnv,
                        const float* lnb, const float* lng,
                        const float* ewb, const float* eww, float* h){
  int t = blockIdx.x; int b = t & (BDIM-1); int k = t >> 6;
  int e = threadIdx.x;
  float mf = mask[b*KDIM+k] ? 1.f : 0.f;
  __shared__ float sh[18]; __shared__ float lnv[18]; __shared__ float stat[2];
  if(e<18){
    float v;
    if(e<16) v = feat[(size_t)b*INF_*KDIM + e*KDIM + k];
    else if(e==16) v = (mf>0.f) ? s1[b*KDIM+k] : 0.f;
    else v = pp[0*NTOK + b*KDIM + k] / sumpt[b];
    v *= mf;
    v = (v - bnm[e]) * rsqrtf(bnv[e]+1e-5f) * bng[e] + bnb[e];
    sh[e]=v;
  }
  __syncthreads();
  if(e==0){
    float mu=0.f; for(int c=0;c<18;++c) mu+=sh[c]; mu*= (1.f/18.f);
    float va=0.f; for(int c=0;c<18;++c){ float d=sh[c]-mu; va+=d*d; } va*=(1.f/18.f);
    stat[0]=mu; stat[1]=rsqrtf(va+1e-5f);
  }
  __syncthreads();
  if(e<18) lnv[e]=(sh[e]-stat[0])*stat[1]*lng[e]+lnb[e];
  __syncthreads();
  float acc = ewb[e];
  for(int c=0;c<18;++c) acc += lnv[c]*eww[e*18+c];
  h[(size_t)t*EDIM + e] = gelu_exact(acc)*mf;
}

// ---------------- LayerNorm (D = 128 or 512) -> f16 output ----------------
__global__ void k_ln(const float* x, const float* g, const float* b, _Float16* y, int D){
  int row = blockIdx.x, tid = threadIdx.x;
  int per = D >> 7;
  float lv[4];
  const float* xr = x + (size_t)row*D;
  float s=0.f;
  for(int q=0;q<per;++q){ lv[q]=xr[tid+(q<<7)]; s+=lv[q]; }
  __shared__ float red[128]; __shared__ float stat[2];
  red[tid]=s; __syncthreads();
  for(int st=64;st>0;st>>=1){ if(tid<st) red[tid]+=red[tid+st]; __syncthreads(); }
  if(tid==0) stat[0]=red[0]/(float)D;
  __syncthreads();
  float mu=stat[0]; s=0.f;
  for(int q=0;q<per;++q){ float d=lv[q]-mu; s+=d*d; }
  red[tid]=s; __syncthreads();
  for(int st=64;st>0;st>>=1){ if(tid<st) red[tid]+=red[tid+st]; __syncthreads(); }
  if(tid==0) stat[1]=rsqrtf(red[0]/(float)D + 1e-5f);
  __syncthreads();
  float rs=stat[1];
  _Float16* yr = y + (size_t)row*D;
  for(int q=0;q<per;++q){ int c=tid+(q<<7); yr[c]=(_Float16)((lv[q]-mu)*rs*g[c]+b[c]); }
}

// ---------------- f32 -> f16, vectorized (n4 = n/4 chunks) ----------------
__global__ void k_f2h(const float* x, _Float16* y, int n4){
  int i = blockIdx.x*256 + threadIdx.x;
  if(i<n4) ((h4*)y)[i] = cvt4(((const f4*)x)[i]);
}

// ---------------- generic WMMA GEMM: C[M,N] = act(A f16 [M,Kd] * W f16 [N,Kd]^T + bias) + resid*rscale ----------------
__global__ __launch_bounds__(256) void k_gemm(const _Float16* A, const _Float16* W, const float* bias,
                       const float* resid, const float* rscale, float* C,
                       int M, int N, int Kd, int act){
  __shared__ _Float16 As[128*40];
  __shared__ _Float16 Bs[64*40];
  int m0 = blockIdx.x*128, n0 = blockIdx.y*64;
  int tid = threadIdx.x, lane = tid&31, wid = tid>>5, hi = lane>>4;
  v8f acc[4] = {};
  for(int kb=0; kb<Kd; kb+=32){
    __syncthreads();
    // async DMA tiles into LDS: 16B per lane (rows of 32 halves = 4 chunks)
#pragma unroll
    for(int q=0;q<2;++q){
      int idx = tid + q*256;             // 512 chunks for A (128 rows x 4)
      int r = idx>>2, c4 = idx&3;
      async_g2l_b128(&As[r*40 + c4*8], &A[(size_t)(m0+r)*Kd + kb + c4*8]);
    }
    {
      int r = tid>>2, c4 = tid&3;        // 256 chunks for B (64 rows x 4)
      async_g2l_b128(&Bs[r*40 + c4*8], &W[(size_t)(n0+r)*Kd + kb + c4*8]);
    }
    wait_async0();
    __syncthreads();
    v16h a = frag_a_rm(As, 40, wid*16, 0);
#pragma unroll
    for(int nt=0;nt<4;++nt){
      v16h bf = frag_b_nk(Bs, 40, nt*16, 0);
      acc[nt] = wmma_f16(a, bf, acc[nt]);
    }
  }
#pragma unroll
  for(int nt=0;nt<4;++nt){
#pragma unroll
    for(int r=0;r<8;++r){
      int row = m0 + wid*16 + r + (hi?8:0);
      int col = n0 + nt*16 + (lane&15);
      float v = acc[nt][r];
      if(bias) v += bias[col];
      if(act==1) v = gelu_exact(v);
      if(resid) v += resid[(size_t)row*N + col] * (rscale ? rscale[col] : 1.f);
      C[(size_t)row*N + col] = v;
    }
  }
}

// ---------------- pairwise bias: features -> bn0 -> conv1+bn1+gelu -> conv2(WMMA)+bn2+gelu -> conv3+bn3 ----------------
__global__ __launch_bounds__(256) void k_pairbias(const float* pp,
    const float* bn0b,const float* bn0g,const float* bn0m,const float* bn0v,
    const float* c1b,const float* c1w,
    const float* bn1b,const float* bn1g,const float* bn1m,const float* bn1v,
    const float* c2b,const float* c2w,
    const float* bn2b,const float* bn2g,const float* bn2m,const float* bn2v,
    const float* c3b,const float* c3w,
    const float* bn3b,const float* bn3g,const float* bn3m,const float* bn3v,
    float* biasout){
  int b   = blockIdx.x >> 4;
  int seg = blockIdx.x & 15;
  int tid = threadIdx.x, lane = tid&31, wid = tid>>5, hi = lane>>4, m = lane&15;
  __shared__ float part[7][128];
  __shared__ _Float16 w2l[64*64];
  __shared__ float w1l[256];
  __shared__ float w3l[256];
  __shared__ float S0[4],T0[4],S3[4],T3[4];
  __shared__ float S1[64],T1[64],S2[64],T2[64];
  for(int idx=tid; idx<7*128; idx+=256) part[idx>>7][idx&127]=pp[(idx>>7)*NTOK + b*KDIM + (idx&127)];
#pragma unroll
  for(int q=0;q<4;++q){                       // 4096 f32 -> f16, 4-wide
    int i4 = tid + q*256;
    ((h4*)w2l)[i4] = cvt4(((const f4*)c2w)[i4]);
  }
  w1l[tid]=c1w[tid];
  w3l[tid]=c3w[tid];
  if(tid<64){
    float rs=rsqrtf(bn1v[tid]+1e-5f); S1[tid]=rs*bn1g[tid]; T1[tid]=(c1b[tid]-bn1m[tid])*S1[tid]+bn1b[tid];
    rs=rsqrtf(bn2v[tid]+1e-5f);       S2[tid]=rs*bn2g[tid]; T2[tid]=(c2b[tid]-bn2m[tid])*S2[tid]+bn2b[tid];
  }
  if(tid<4){
    float rs=rsqrtf(bn0v[tid]+1e-5f); S0[tid]=rs*bn0g[tid]; T0[tid]=bn0b[tid]-bn0m[tid]*S0[tid];
    rs=rsqrtf(bn3v[tid]+1e-5f);       S3[tid]=rs*bn3g[tid]; T3[tid]=(c3b[tid]-bn3m[tid])*S3[tid]+bn3b[tid];
  }
  __syncthreads();
  v16h bf[4][2];
#pragma unroll
  for(int nt=0;nt<4;++nt)
#pragma unroll
    for(int kc=0;kc<2;++kc)
      bf[nt][kc]=frag_b_nk(w2l, 64, nt*16, kc*32);
  for(int it=0; it<8; ++it){
    int tile = seg*64 + wid*8 + it;   // 1024 tiles of 16 pairs per batch element
    int p0 = tile*16;
    int i  = p0 >> 7;
    int j0 = p0 & 127;
    int j  = j0 + m;
    float pti=part[0][i], rapi=part[1][i], phii=part[2][i], pxi=part[3][i], pyi=part[4][i], pzi=part[5][i], ei=part[6][i];
    float ptj=part[0][j], rapj=part[1][j], phij=part[2][j], pxj=part[3][j], pyj=part[4][j], pzj=part[5][j], ej=part[6][j];
    float drap = rapi-rapj;
    const float PI_ = 3.14159265358979323846f, PI2 = 6.28318530717958647692f;
    float dp = phii-phij + PI_;
    dp = dp - floorf(dp/PI2)*PI2 - PI_;
    float delta = sqrtf(drap*drap + dp*dp);
    float ptmin = fminf(pti,ptj);
    float f[4];
    f[0] = logf(fmaxf(ptmin*delta, 1e-8f));                       // lnkt
    f[1] = logf(fmaxf(ptmin / fmaxf(pti+ptj,1e-8f), 1e-8f));      // lnz
    f[2] = logf(fmaxf(delta, 1e-8f));                             // lndelta
    float se=ei+ej, sx=pxi+pxj, sy=pyi+pyj, sz=pzi+pzj;
    f[3] = logf(fmaxf(se*se - sx*sx - sy*sy - sz*sz, 1e-8f));     // lnm2
#pragma unroll
    for(int c=0;c<4;++c) f[c]=f[c]*S0[c]+T0[c];
    // conv1 + bn1 + gelu computed directly in WMMA A-fragment channel order
    v16h af[2];
#pragma unroll
    for(int kc=0;kc<2;++kc){
#pragma unroll
      for(int t=0;t<16;++t){
        int ch = kc*32 + (hi?16:0) + t;
        float conv = f[0]*w1l[ch*4+0]+f[1]*w1l[ch*4+1]+f[2]*w1l[ch*4+2]+f[3]*w1l[ch*4+3];
        af[kc][t]=(_Float16)gelu_exact(conv*S1[ch]+T1[ch]);
      }
    }
    v8f acc[4] = {};
#pragma unroll
    for(int kc=0;kc<2;++kc)
#pragma unroll
      for(int nt=0;nt<4;++nt)
        acc[nt]=wmma_f16(af[kc], bf[nt][kc], acc[nt]);
    // conv3 + bn3 with cross-lane channel reduction (16-lane halves)
#pragma unroll
    for(int r=0;r<8;++r){
      int mm = r + (hi?8:0);
      float s[4]={0.f,0.f,0.f,0.f};
#pragma unroll
      for(int nt=0;nt<4;++nt){
        int ch = nt*16 + m;
        float a2 = gelu_exact(acc[nt][r]*S2[ch]+T2[ch]);
        s[0]+=a2*w3l[ch]; s[1]+=a2*w3l[64+ch]; s[2]+=a2*w3l[128+ch]; s[3]+=a2*w3l[192+ch];
      }
#pragma unroll
      for(int hh=0;hh<4;++hh)
        for(int off=8; off>0; off>>=1) s[hh]+=__shfl_xor(s[hh],off,32);
      if(m<4)
        biasout[(size_t)((b*HN+m)*KDIM + i)*KDIM + (j0+mm)] = s[m]*S3[m]+T3[m];
    }
  }
}

// ---------------- attention per (b, h, 32-query tile) ----------------
__global__ __launch_bounds__(256) void k_attn(const float* qkv, const float* biasT,
                                              const unsigned char* mask, float* o){
  int blk = blockIdx.x;
  int q0 = (blk & 3) * 32;
  int bh = blk >> 2;
  int b = bh >> 2, h = bh & 3;
  int tid = threadIdx.x, lane = tid&31, wid = tid>>5, hi = lane>>4;
  __shared__ _Float16 qs[32*40];
  __shared__ _Float16 ks[128*40];
  __shared__ _Float16 vs[128*40];
  __shared__ float    sc[32*132];
  __shared__ _Float16 at[32*136];
  __shared__ unsigned char mk[128];
  if(tid<128) mk[tid]=mask[b*KDIM+tid];
  // stage q/k/v tiles as f16 (vectorized: float4 load -> 4xf16 store)
  if(tid < 32*8){ int r=tid>>3, d4=tid&7;
    *(h4*)&qs[r*40+d4*4] = cvt4(*(const f4*)&qkv[(size_t)((q0+r)*BDIM+b)*384 + h*32 + d4*4]); }
#pragma unroll
  for(int q=0;q<4;++q){ int idx = tid + q*256; int r=idx>>3, d4=idx&7;
    *(h4*)&ks[r*40+d4*4] = cvt4(*(const f4*)&qkv[(size_t)(r*BDIM+b)*384 + 128 + h*32 + d4*4]);
    *(h4*)&vs[r*40+d4*4] = cvt4(*(const f4*)&qkv[(size_t)(r*BDIM+b)*384 + 256 + h*32 + d4*4]); }
  __syncthreads();
  { // scores = q k^T / sqrt(HD) + bias, pad -> -inf
    int mt  = wid >> 2;
    int ntb = (wid & 3) * 2;
    const float scale = 0.1767766952966369f; // 1/sqrt(32)
    v16h a = frag_a_rm(qs, 40, mt*16, 0);
#pragma unroll
    for(int u=0;u<2;++u){
      int nt = ntb+u;
      v16h bfk = frag_b_nk(ks, 40, nt*16, 0);
      v8f accv = {};
      accv = wmma_f16(a, bfk, accv);
#pragma unroll
      for(int r=0;r<8;++r){
        int qr = mt*16 + r + (hi?8:0);
        int j  = nt*16 + (lane&15);
        float v = accv[r]*scale + biasT[(size_t)((b*HN+h)*KDIM + (q0+qr))*KDIM + j];
        if(!mk[j]) v = -INFINITY;
        sc[qr*132 + j] = v;
      }
    }
  }
  __syncthreads();
  { // softmax over rows (4 rows per wave)
#pragma unroll
    for(int rr=0; rr<4; ++rr){
      int row = wid*4 + rr;
      float v[4], mx=-INFINITY;
#pragma unroll
      for(int q=0;q<4;++q){ v[q]=sc[row*132 + lane + q*32]; mx=fmaxf(mx,v[q]); }
      for(int off=16; off>0; off>>=1) mx=fmaxf(mx, __shfl_xor(mx,off,32));
      float e[4], s=0.f;
#pragma unroll
      for(int q=0;q<4;++q){ float ev = (v[q] > -1e37f) ? __expf(v[q]-mx) : 0.f; e[q]=ev; s+=ev; }
      for(int off=16; off>0; off>>=1) s += __shfl_xor(s,off,32);
      float inv = (s>0.f) ? 1.f/s : 0.f;
#pragma unroll
      for(int q=0;q<4;++q) at[row*136 + lane + q*32] = (_Float16)(e[q]*inv);
    }
  }
  __syncthreads();
  if(wid < 4){ // attn @ V
    int mt = wid>>1, nt = wid&1;
    v8f accv = {};
#pragma unroll
    for(int kc=0;kc<4;++kc){
      v16h a  = frag_a_rm(at, 136, mt*16, kc*32);
      v16h bfv = frag_b_kn(vs, 40, nt*16, kc*32);
      accv = wmma_f16(a, bfv, accv);
    }
#pragma unroll
    for(int r=0;r<8;++r){
      int qr = q0 + mt*16 + r + (hi?8:0);
      int d  = nt*16 + (lane&15);
      o[(size_t)(qr*BDIM+b)*EDIM + h*32 + d] = accv[r];
    }
  }
}

// ---------------- c_attn permuted scale + post-attn LN + residual add (h updated in place) ----------------
__global__ void k_postattn(const float* ao, const float* catt, const float* g, const float* bb, float* h){
  int t = blockIdx.x, e = threadIdx.x;
  int hh = e & 3, d = e >> 2;                     // out[e = d*H + hh] = ao[hh*HD + d] * c_attn[hh]
  float v = ao[(size_t)t*EDIM + hh*32 + d] * catt[hh];
  float res = h[(size_t)t*EDIM + e];
  __shared__ float red[128]; __shared__ float stat[2];
  red[e]=v; __syncthreads();
  for(int s=64;s>0;s>>=1){ if(e<s) red[e]+=red[e+s]; __syncthreads(); }
  if(e==0) stat[0]=red[0]*(1.f/128.f);
  __syncthreads();
  float mu=stat[0]; float dd=v-mu;
  red[e]=dd*dd; __syncthreads();
  for(int s=64;s>0;s>>=1){ if(e<s) red[e]+=red[e+s]; __syncthreads(); }
  if(e==0) stat[1]=rsqrtf(red[0]*(1.f/128.f)+1e-5f);
  __syncthreads();
  h[(size_t)t*EDIM+e] = dd*stat[1]*g[e]+bb[e] + res;
}

// ---------------- head2 (E->1) + pad mask ----------------
__global__ void k_head2(const float* s1, const float* w, const float* b2,
                        const unsigned char* mask, float* out){
  int t = blockIdx.x*256 + threadIdx.x;
  int b = t & 63, k = t >> 6;
  float acc = b2[0];
  const f4* r = (const f4*)(s1 + (size_t)t*EDIM);
  const f4* wv = (const f4*)w;
  for(int c=0;c<EDIM/4;++c){ f4 a = r[c], ww = wv[c]; acc += a[0]*ww[0]+a[1]*ww[1]+a[2]*ww[2]+a[3]*ww[3]; }
  out[b*KDIM + k] = mask[b*KDIM+k] ? acc : -INFINITY;
}

// =====================================================================
extern "C" void kernel_launch(void* const* d_in, const int* in_sizes, int n_in,
                              void* d_out, int out_size, void* d_ws, size_t ws_size,
                              hipStream_t stream){
  (void)in_sizes; (void)n_in; (void)out_size; (void)ws_size;
  const float* feat = (const float*)d_in[1];
  const float* lv   = (const float*)d_in[2];
  const unsigned char* mask = (const unsigned char*)d_in[3];
  const float* s1sc = (const float*)d_in[4];
  auto F = [&](int i){ return (const float*)d_in[i]; };
  // jax-sorted param leaf order: blocks(3 x 18 leaves) then embed_bn(b,g,m,v), embed_ln(b,g),
  // embed_lin(b,w), head1(b,w), head2(b,w), out_ln(b,g), pair(bn0..bn3 b/g/m/v, conv1 b/w, conv2 b/w, conv3 b/w)
  const int PB = 5;

  char* ws = (char*)d_ws;
  size_t off = 0;
  auto alloc = [&](size_t bytes)->void*{ void* p = ws + off; off += (bytes + 255) & ~(size_t)255; return p; };
  float*    PP    = (float*)alloc((size_t)7*NTOK*4);
  float*    SUMPT = (float*)alloc(BDIM*4);
  float*    Hb    = (float*)alloc((size_t)NTOK*EDIM*4);
  _Float16* XLN   = (_Float16*)alloc((size_t)NTOK*EDIM*2);
  float*    QKV   = (float*)alloc((size_t)NTOK*FFND*4);     // also reused as fc1 output
  _Float16* LN512 = (_Float16*)alloc((size_t)NTOK*FFND*2);
  float*    O     = (float*)alloc((size_t)NTOK*EDIM*4);
  _Float16* O16   = (_Float16*)alloc((size_t)NTOK*EDIM*2);
  float*    AO    = (float*)alloc((size_t)NTOK*EDIM*4);     // also reused as head1 output
  float*    BIAS  = (float*)alloc((size_t)BDIM*HN*KDIM*KDIM*4);
  _Float16* W16[NL_][4];
  for(int l=0;l<NL_;++l){
    W16[l][0]=(_Float16*)alloc(384*128*2);
    W16[l][1]=(_Float16*)alloc(128*128*2);
    W16[l][2]=(_Float16*)alloc(512*128*2);
    W16[l][3]=(_Float16*)alloc(128*512*2);
  }
  _Float16* WH1 = (_Float16*)alloc(128*128*2);

  for(int l=0;l<NL_;++l){
    k_f2h<<<dim3(384*128/4/256),256,0,stream>>>(F(PB+l*18+1), W16[l][0], 384*128/4);
    k_f2h<<<dim3(128*128/4/256),256,0,stream>>>(F(PB+l*18+3), W16[l][1], 128*128/4);
    k_f2h<<<dim3(512*128/4/256),256,0,stream>>>(F(PB+l*18+6), W16[l][2], 512*128/4);
    k_f2h<<<dim3(128*512/4/256),256,0,stream>>>(F(PB+l*18+8), W16[l][3], 128*512/4);
  }
  k_f2h<<<dim3(128*128/4/256),256,0,stream>>>(F(68), WH1, 128*128/4);

  k_particle<<<BDIM,KDIM,0,stream>>>(lv, mask, PP, SUMPT);
  k_embed<<<NTOK,128,0,stream>>>(feat, s1sc, mask, PP, SUMPT,
      F(59),F(60),F(61),F(62), F(63),F(64), F(65),F(66), Hb);
  k_pairbias<<<BDIM*16,256,0,stream>>>(PP,
      F(73),F(74),F(75),F(76),          // bn0 b,g,m,v
      F(89),F(90),                      // conv1 b,w
      F(77),F(78),F(79),F(80),          // bn1
      F(91),F(92),                      // conv2 b,w
      F(81),F(82),F(83),F(84),          // bn2
      F(93),F(94),                      // conv3 b,w
      F(85),F(86),F(87),F(88),          // bn3
      BIAS);

  for(int l=0;l<NL_;++l){
    const float* AIb = F(PB+l*18+0);
    const float* AOb = F(PB+l*18+2);
    const float* CAT = F(PB+l*18+4);
    const float* F1b = F(PB+l*18+5);
    const float* F2b = F(PB+l*18+7);
    const float* LPAb= F(PB+l*18+9),  *LPAg= F(PB+l*18+10);
    const float* LPFb= F(PB+l*18+11), *LPFg= F(PB+l*18+12);
    const float* LRAb= F(PB+l*18+13), *LRAg= F(PB+l*18+14);
    const float* LRFb= F(PB+l*18+15), *LRFg= F(PB+l*18+16);
    const float* WRES= F(PB+l*18+17);
    k_ln<<<NTOK,128,0,stream>>>(Hb, LRAg, LRAb, XLN, 128);
    k_gemm<<<dim3(64,6),256,0,stream>>>(XLN, W16[l][0], AIb, nullptr,nullptr, QKV, NTOK,384,128, 0);
    k_attn<<<BDIM*HN*4,256,0,stream>>>(QKV, BIAS, mask, O);
    k_f2h<<<dim3(NTOK*EDIM/4/256),256,0,stream>>>(O, O16, NTOK*EDIM/4);
    k_gemm<<<dim3(64,2),256,0,stream>>>(O16, W16[l][1], AOb, nullptr,nullptr, AO, NTOK,128,128, 0);
    k_postattn<<<NTOK,128,0,stream>>>(AO, CAT, LPAg, LPAb, Hb);
    k_ln<<<NTOK,128,0,stream>>>(Hb, LRFg, LRFb, XLN, 128);
    k_gemm<<<dim3(64,8),256,0,stream>>>(XLN, W16[l][2], F1b, nullptr,nullptr, QKV, NTOK,512,128, 1);
    k_ln<<<NTOK,128,0,stream>>>(QKV, LPFg, LPFb, LN512, 512);
    k_gemm<<<dim3(64,2),256,0,stream>>>(LN512, W16[l][3], F2b, Hb, WRES, Hb, NTOK,128,512, 0);
  }

  k_ln<<<NTOK,128,0,stream>>>(Hb, F(72), F(71), XLN, 128);
  k_gemm<<<dim3(64,2),256,0,stream>>>(XLN, WH1, F(67), nullptr,nullptr, AO, NTOK,128,128, 1);
  k_head2<<<NTOK/256,256,0,stream>>>(AO, F(70), F(69), mask, (float*)d_out);
}
